// GNN_69166153335013
// MI455X (gfx1250) — compile-verified, matrix operand order checked by reading, not run on hardware
//
#include <hip/hip_runtime.h>
#include <hip/hip_bf16.h>

#define NNODES 100000
#define NEDGES 1600000
#define DIM    64
#define DIM2   128
#define NGRAPH 256
#define SCANB  1024
#define NSCANBLK ((NNODES + SCANB - 1) / SCANB)   // 98

typedef __attribute__((ext_vector_type(16))) __bf16 v16bf;
typedef __attribute__((ext_vector_type(8)))  float  v8f;

union Frag16 {
    v16bf          v;
    unsigned short s[16];
    unsigned int   u[8];
};

__device__ __forceinline__ unsigned short f32_to_bf16(float f) {
    unsigned int u = __float_as_uint(f);
    unsigned int r = u + 0x7FFFu + ((u >> 16) & 1u);   // round-to-nearest-even
    return (unsigned short)(r >> 16);
}

// ---------------------------------------------------------------- utilities
__global__ void zero_i32(int* __restrict__ p, int n) {
    int i = blockIdx.x * blockDim.x + threadIdx.x;
    if (i < n) p[i] = 0;
}

__global__ void cvt_bf16(const float* __restrict__ in, unsigned short* __restrict__ outp, int n) {
    int i = blockIdx.x * blockDim.x + threadIdx.x;
    if (i < n) outp[i] = f32_to_bf16(in[i]);
}

// ------------------------------------------------------------- CSR by dst
// 1) degree histogram (int atomics, cheap)
__global__ __launch_bounds__(256) void csr_hist(const int* __restrict__ ei,
                                                int* __restrict__ deg) {
    int e = blockIdx.x * blockDim.x + threadIdx.x;
    if (e >= NEDGES) return;
    atomicAdd(&deg[ei[NEDGES + e]], 1);
}

// 2a) per-1024-block inclusive scan of degrees
__global__ __launch_bounds__(SCANB) void scan_block(const int* __restrict__ deg,
                                                    int* __restrict__ incl,
                                                    int* __restrict__ blkSums) {
    __shared__ int buf[SCANB];
    int gid = blockIdx.x * SCANB + threadIdx.x;
    int v = (gid < NNODES) ? deg[gid] : 0;
    buf[threadIdx.x] = v;
    __syncthreads();
    for (int offs = 1; offs < SCANB; offs <<= 1) {
        int t = (threadIdx.x >= offs) ? buf[threadIdx.x - offs] : 0;
        __syncthreads();
        buf[threadIdx.x] += t;
        __syncthreads();
    }
    if (gid < NNODES) incl[gid] = buf[threadIdx.x];
    if (threadIdx.x == SCANB - 1) blkSums[blockIdx.x] = buf[SCANB - 1];
}

// 2b) single-block exclusive scan of the 98 block sums (in place)
__global__ __launch_bounds__(128) void scan_sums(int* __restrict__ blkSums, int nblk) {
    __shared__ int buf[128];
    int v = (threadIdx.x < nblk) ? blkSums[threadIdx.x] : 0;
    buf[threadIdx.x] = v;
    __syncthreads();
    for (int offs = 1; offs < 128; offs <<= 1) {
        int t = (threadIdx.x >= offs) ? buf[threadIdx.x - offs] : 0;
        __syncthreads();
        buf[threadIdx.x] += t;
        __syncthreads();
    }
    if (threadIdx.x < nblk) blkSums[threadIdx.x] = buf[threadIdx.x] - v;   // exclusive
}

// 2c) rowstart = exclusive scan; cursor = copy; rowstart[N] = E
__global__ void scan_finalize(const int* __restrict__ incl, const int* __restrict__ deg,
                              const int* __restrict__ blkOffs,
                              int* __restrict__ rowstart, int* __restrict__ cursor) {
    int gid = blockIdx.x * blockDim.x + threadIdx.x;
    if (gid < NNODES) {
        int ex = incl[gid] - deg[gid] + blkOffs[gid >> 10];   // >>10 matches SCANB=1024
        rowstart[gid] = ex;
        cursor[gid]   = ex;
    }
    if (gid == 0) rowstart[NNODES] = NEDGES;
}

// 3) slot fill: csrc[slot] = src for each in-edge of dst
__global__ __launch_bounds__(256) void csr_fill(const int* __restrict__ ei,
                                                int* __restrict__ cursor,
                                                int* __restrict__ csrc) {
    int e = blockIdx.x * blockDim.x + threadIdx.x;
    if (e >= NEDGES) return;
    int s = ei[e];
    int d = ei[NEDGES + e];
    int slot = atomicAdd(&cursor[d], 1);
    csrc[slot] = s;
}

// ------------------------------------------------ gather-style aggregation
// One wave per destination node; lane l accumulates dims l and l+32 over the
// in-edge list. Each source row is one exact 256B coalesced L2 read; no
// float atomics, no zero-init pass.
__global__ __launch_bounds__(256) void agg_gather(const float* __restrict__ x,
                                                  const int* __restrict__ rowstart,
                                                  const int* __restrict__ csrc,
                                                  float* __restrict__ agg) {
    int node = (blockIdx.x * 256 + threadIdx.x) >> 5;
    int lane = threadIdx.x & 31;
    if (node >= NNODES) return;
    int lo = rowstart[node], hi = rowstart[node + 1];
    float a0 = 0.0f, a1 = 0.0f;
    for (int k = lo; k < hi; ++k) {
        const float* xr = x + (size_t)csrc[k] * DIM;
        a0 += xr[lane];
        a1 += xr[lane + 32];
    }
    agg[(size_t)node * DIM + lane]      = a0;
    agg[(size_t)node * DIM + lane + 32] = a1;
}

// --------------------------------------------------- GEMM1: [N,64]x[64,128]
// out = tanh( ((1+eps)*x + agg) @ W1 * bn_scale + bn_shift ), bf16 output.
#define LDA1 66   // padded bf16 row stride (64+2) -> conflict-free LDS
__global__ __launch_bounds__(256) void gin_gemm1(
    const float* __restrict__ x, const float* __restrict__ agg,
    const float* __restrict__ eps_p,
    const unsigned short* __restrict__ Wbf,   // [64][128] bf16 row-major
    const float* __restrict__ bias, const float* __restrict__ gamma,
    const float* __restrict__ beta,
    unsigned short* __restrict__ out)         // [N][128] bf16
{
    __shared__ unsigned short ldsA[16 * LDA1];
    const int rowBlk = blockIdx.x << 4;
    const float ep = 1.0f + eps_p[0];

    for (int i = threadIdx.x; i < 16 * DIM; i += 256) {
        const int r = i >> 6, c = i & 63;
        const int g = (rowBlk + r) * DIM + c;
        ldsA[r * LDA1 + c] = f32_to_bf16(fmaf(ep, x[g], agg[g]));
    }
    __syncthreads();

    const int lane = threadIdx.x & 31;
    const int wv   = threadIdx.x >> 5;   // n-tile 0..7
    const int hl   = lane >> 4;          // half-wave K split (ISA 16-bit layout)
    const int m    = lane & 15;
    const int ncol = (wv << 4) | m;

    v8f acc = {0.f, 0.f, 0.f, 0.f, 0.f, 0.f, 0.f, 0.f};
#pragma unroll
    for (int k0 = 0; k0 < DIM; k0 += 32) {
        Frag16 a, b;
#pragma unroll
        for (int j = 0; j < 8; ++j) {
            const int kb = ((j < 4) ? (2 * j) : (16 + 2 * (j - 4))) + 8 * hl;
            a.u[j] = *(const unsigned int*)&ldsA[m * LDA1 + k0 + kb];
            b.s[2 * j]     = Wbf[(k0 + kb)     * DIM2 + ncol];
            b.s[2 * j + 1] = Wbf[(k0 + kb + 1) * DIM2 + ncol];
        }
        acc = __builtin_amdgcn_wmma_f32_16x16x32_bf16(false, a.v, false, b.v,
                                                      (short)0, acc, false, false);
    }
    const float sc = gamma[ncol] * rsqrtf(1.0f + 1e-5f);   // BN eval scale
    const float sh = fmaf(bias[ncol], sc, beta[ncol]);     // fold bias + beta
#pragma unroll
    for (int r = 0; r < 8; ++r) {
        const int row = rowBlk + r + 8 * hl;
        out[(size_t)row * DIM2 + ncol] = f32_to_bf16(tanhf(fmaf(acc[r], sc, sh)));
    }
}

// --------------------------------------------------- GEMM2: [N,128]x[128,64]
// out = tanh( t @ W2 + b2 ), f32 output (feeds next layer / pooling).
#define LDA2 130
__global__ __launch_bounds__(128) void gin_gemm2(
    const unsigned short* __restrict__ tin,   // [N][128] bf16
    const unsigned short* __restrict__ Wbf,   // [128][64] bf16 row-major
    const float* __restrict__ bias,           // [64]
    float* __restrict__ out)                  // [N][64] f32
{
    __shared__ unsigned short ldsA[16 * LDA2];
    const int rowBlk = blockIdx.x << 4;

    for (int i = threadIdx.x; i < 16 * 64; i += 128) {    // 64 bf16-pair cols
        const int r = i >> 6, cp = i & 63;
        const unsigned int v = ((const unsigned int*)(tin + (size_t)(rowBlk + r) * DIM2))[cp];
        *(unsigned int*)&ldsA[r * LDA2 + 2 * cp] = v;
    }
    __syncthreads();

    const int lane = threadIdx.x & 31;
    const int wv   = threadIdx.x >> 5;   // n-tile 0..3
    const int hl   = lane >> 4;
    const int m    = lane & 15;
    const int ncol = (wv << 4) | m;

    v8f acc = {0.f, 0.f, 0.f, 0.f, 0.f, 0.f, 0.f, 0.f};
#pragma unroll
    for (int k0 = 0; k0 < DIM2; k0 += 32) {
        Frag16 a, b;
#pragma unroll
        for (int j = 0; j < 8; ++j) {
            const int kb = ((j < 4) ? (2 * j) : (16 + 2 * (j - 4))) + 8 * hl;
            a.u[j] = *(const unsigned int*)&ldsA[m * LDA2 + k0 + kb];
            b.s[2 * j]     = Wbf[(k0 + kb)     * DIM + ncol];
            b.s[2 * j + 1] = Wbf[(k0 + kb + 1) * DIM + ncol];
        }
        acc = __builtin_amdgcn_wmma_f32_16x16x32_bf16(false, a.v, false, b.v,
                                                      (short)0, acc, false, false);
    }
#pragma unroll
    for (int r = 0; r < 8; ++r) {
        const int row = rowBlk + r + 8 * hl;
        out[(size_t)row * DIM + ncol] = tanhf(acc[r] + bias[ncol]);
    }
}

// -------------------------------- fused mean-pool (sorted batch) + linear
// batch is sorted ascending, so each graph is a contiguous node range found
// by binary search. No atomics; fully deterministic.
__global__ __launch_bounds__(64) void pool_final(const float* __restrict__ h,
                                                 const int* __restrict__ batch,
                                                 const float* __restrict__ linW,
                                                 const float* __restrict__ linb,
                                                 float* __restrict__ outp) {
    __shared__ float pooled[DIM];
    __shared__ int bounds[2];
    const int g = blockIdx.x;
    if (threadIdx.x < 2) {
        int target = g + threadIdx.x;        // lower_bound(batch, target)
        int lo = 0, hi = NNODES;
        while (lo < hi) {
            int mid = (lo + hi) >> 1;
            if (batch[mid] < target) lo = mid + 1; else hi = mid;
        }
        bounds[threadIdx.x] = lo;
    }
    __syncthreads();
    const int lo = bounds[0], hi = bounds[1];
    const int n = threadIdx.x;
    float acc = 0.0f;
    for (int r = lo; r < hi; ++r) acc += h[(size_t)r * DIM + n];
    pooled[n] = acc / fmaxf((float)(hi - lo), 1.0f);
    __syncthreads();
    float a = 0.0f;
#pragma unroll 8
    for (int k = 0; k < DIM; ++k) a = fmaf(pooled[k], linW[k * DIM + n], a);
    outp[g * DIM + n] = tanhf(a + linb[n]);
}

// ---------------------------------------------------------------- launcher
extern "C" void kernel_launch(void* const* d_in, const int* in_sizes, int n_in,
                              void* d_out, int out_size, void* d_ws, size_t ws_size,
                              hipStream_t stream) {
    (void)in_sizes; (void)n_in; (void)out_size; (void)ws_size;
    const float* x      = (const float*)d_in[0];
    const int*   ei     = (const int*)d_in[1];
    const int*   batch  = (const int*)d_in[2];
    const float* eps1   = (const float*)d_in[3];
    const float* W11    = (const float*)d_in[4];
    const float* b11    = (const float*)d_in[5];
    const float* gamma1 = (const float*)d_in[6];
    const float* beta1  = (const float*)d_in[7];
    const float* W12    = (const float*)d_in[8];
    const float* b12    = (const float*)d_in[9];
    const float* eps2   = (const float*)d_in[10];
    const float* W21    = (const float*)d_in[11];
    const float* b21    = (const float*)d_in[12];
    const float* gamma2 = (const float*)d_in[13];
    const float* beta2  = (const float*)d_in[14];
    const float* W22    = (const float*)d_in[15];
    const float* b22    = (const float*)d_in[16];
    const float* linW   = (const float*)d_in[17];
    const float* linb   = (const float*)d_in[18];
    float* outp = (float*)d_out;

    char* ws = (char*)d_ws;
    size_t off = 0;
    auto carve = [&](size_t bytes) -> char* {
        char* p = ws + off;
        off = (off + bytes + 255) & ~(size_t)255;
        return p;
    };
    float*          agg      = (float*)carve((size_t)NNODES * DIM * 4);
    unsigned short* tbuf     = (unsigned short*)carve((size_t)NNODES * DIM2 * 2);
    float*          h1       = (float*)carve((size_t)NNODES * DIM * 4);
    int*            csrc     = (int*)carve((size_t)NEDGES * 4);
    int*            deg      = (int*)carve((size_t)NNODES * 4);
    int*            incl     = (int*)carve((size_t)NNODES * 4);
    int*            rowstart = (int*)carve((size_t)(NNODES + 1) * 4);
    int*            cursor   = (int*)carve((size_t)NNODES * 4);
    int*            blkSums  = (int*)carve(128 * 4);
    unsigned short* w1b      = (unsigned short*)carve(DIM * DIM2 * 2);
    unsigned short* w2b      = (unsigned short*)carve(DIM2 * DIM * 2);
    unsigned short* w3b      = (unsigned short*)carve(DIM * DIM2 * 2);
    unsigned short* w4b      = (unsigned short*)carve(DIM2 * DIM * 2);

    // weight conversions (f32 -> bf16)
    cvt_bf16<<<(DIM * DIM2 + 255) / 256, 256, 0, stream>>>(W11, w1b, DIM * DIM2);
    cvt_bf16<<<(DIM2 * DIM + 255) / 256, 256, 0, stream>>>(W12, w2b, DIM2 * DIM);
    cvt_bf16<<<(DIM * DIM2 + 255) / 256, 256, 0, stream>>>(W21, w3b, DIM * DIM2);
    cvt_bf16<<<(DIM2 * DIM + 255) / 256, 256, 0, stream>>>(W22, w4b, DIM2 * DIM);

    const int gridE = (NEDGES + 255) / 256;
    const int gridN = (NNODES + 255) / 256;
    const int gridM = NNODES / 16;               // 6250, exact
    const int gridW = (NNODES * 32 + 255) / 256; // one wave per node

    // ----- CSR build (reused by both layers)
    zero_i32<<<gridN, 256, 0, stream>>>(deg, NNODES);
    csr_hist<<<gridE, 256, 0, stream>>>(ei, deg);
    scan_block<<<NSCANBLK, SCANB, 0, stream>>>(deg, incl, blkSums);
    scan_sums<<<1, 128, 0, stream>>>(blkSums, NSCANBLK);
    scan_finalize<<<gridN, 256, 0, stream>>>(incl, deg, blkSums, rowstart, cursor);
    csr_fill<<<gridE, 256, 0, stream>>>(ei, cursor, csrc);

    // ----- layer 1
    agg_gather<<<gridW, 256, 0, stream>>>(x, rowstart, csrc, agg);
    gin_gemm1<<<gridM, 256, 0, stream>>>(x, agg, eps1, w1b, b11, gamma1, beta1, tbuf);
    gin_gemm2<<<gridM, 128, 0, stream>>>(tbuf, w2b, b12, h1);

    // ----- layer 2 (same CSR)
    agg_gather<<<gridW, 256, 0, stream>>>(h1, rowstart, csrc, agg);
    gin_gemm1<<<gridM, 256, 0, stream>>>(h1, agg, eps2, w3b, b21, gamma2, beta2, tbuf);
    gin_gemm2<<<gridM, 128, 0, stream>>>(tbuf, w4b, b22, h1);

    // ----- fused global mean pool + final linear + tanh
    pool_final<<<NGRAPH, 64, 0, stream>>>(h1, batch, linW, linb, outp);
}
